// CvTAttentionBlock_25305947308382
// MI455X (gfx1250) — matrix operand order repaired and checked by reading, not compiled, not run on hardware
//
#include <hip/hip_runtime.h>

// ---- problem constants -------------------------------------------------
#define B_    8
#define HQ_   56
#define WQ_   56
#define HK_   28
#define WK_   28
#define C_    384
#define NH_   6
#define HD_   64
#define NQ_   3136        // 56*56
#define NKV_  784         // 28*28
#define NKVP_ 800         // keys padded to multiple of 32
#define MQ_   (B_*NQ_)    // 25088 = 392*64
#define MKV_  (B_*NKV_)   // 6272  = 98*64
#define KCH_  (C_/32)     // 12 k-chunks

typedef __attribute__((ext_vector_type(16))) _Float16 v16h;
typedef __attribute__((ext_vector_type(8)))  _Float16 v8h;
typedef __attribute__((ext_vector_type(8)))  float    v8f;

// ---- WMMA fragment helpers (CDNA5 16x16x32 f16 layouts) ----------------
static __device__ __forceinline__ v16h frag_pair(v8h lo, v8h hi) {
  return __builtin_shufflevector(lo, hi, 0,1,2,3,4,5,6,7,8,9,10,11,12,13,14,15);
}
// A 16x32: lane holds row (lane&15); elems 0..15 = K base+0..15 (contiguous)
// caller passes p = &X[row*ld + k0 + (lane>=16 ? 16 : 0)]
static __device__ __forceinline__ v16h frag_a16(const _Float16* p) {
  v8h lo = *(const v8h*)p;
  v8h hi = *(const v8h*)(p + 8);
  return frag_pair(lo, hi);
}
// B 32x16 from column-contiguous storage: lane holds col (lane&15),
// 16 contiguous K values; caller passes p = &BT[col*ldb + k0 + (lane>=16 ? 16 : 0)]
static __device__ __forceinline__ v16h frag_b16(const _Float16* p) {
  v8h lo = *(const v8h*)p;
  v8h hi = *(const v8h*)(p + 8);
  return frag_pair(lo, hi);
}
static __device__ __forceinline__ v8f wmma_f16(v16h a, v16h b, v8f c) {
  return __builtin_amdgcn_wmma_f32_16x16x32_f16(false, a, false, b, (short)0, c,
                                                false, false);
}
// async DMA: global -> LDS, 16 bytes per lane, tracked by ASYNCcnt
static __device__ __forceinline__ void async_g2l_b128(unsigned lds_off,
                                                      const _Float16* gaddr) {
  asm volatile("global_load_async_to_lds_b128 %0, %1, off"
               :: "v"(lds_off), "v"(gaddr) : "memory");
}

// ---- 1) depthwise 3x3 conv + BN(eval), SAME padding, f16 output --------
__global__ void dwbn_kernel(const float* __restrict__ x,
                            const float* __restrict__ dw,
                            const float* __restrict__ sc, const float* __restrict__ bi,
                            const float* __restrict__ mn, const float* __restrict__ vr,
                            _Float16* __restrict__ out,
                            int Hin, int Win, int Hout, int Wout, int stride, int pad) {
  long idx = (long)blockIdx.x * blockDim.x + threadIdx.x;
  long total = (long)B_ * Hout * Wout * C_;
  if (idx >= total) return;
  int c = (int)(idx % C_); long t = idx / C_;
  int x0 = (int)(t % Wout); t /= Wout;
  int y0 = (int)(t % Hout); int b = (int)(t / Hout);
  float acc = 0.f;
#pragma unroll
  for (int dy = 0; dy < 3; ++dy) {
    int yi = y0 * stride + dy - pad;
    if (yi < 0 || yi >= Hin) continue;
#pragma unroll
    for (int dx = 0; dx < 3; ++dx) {
      int xi = x0 * stride + dx - pad;
      if (xi < 0 || xi >= Win) continue;
      acc += x[(((long)b * Hin + yi) * Win + xi) * C_ + c] * dw[(dy * 3 + dx) * C_ + c];
    }
  }
  float g = sc[c] * rsqrtf(vr[c] + 1e-5f);
  out[idx] = (_Float16)((acc - mn[c]) * g + bi[c]);
}

// ---- 2) weight transpose f32[IC][OC] -> f16[OC][IC] --------------------
__global__ void wtrans_kernel(const float* __restrict__ w, _Float16* __restrict__ wt) {
  int idx = blockIdx.x * blockDim.x + threadIdx.x;
  if (idx >= C_ * C_) return;
  int o = idx % C_, i = idx / C_;
  wt[o * C_ + i] = (_Float16)w[i * C_ + o];
}

// ---- 3) GEMM: Y[M][384] = X[M][384] * Wt^T -----------------------------
// 128 threads = 4 waves; workgroup tile 64x64. Weight chunk [64 cols][32 k]
// (4 KB) is staged once per workgroup into a double-buffered LDS tile via
// async DMA, overlapping the next chunk's transfer with this chunk's WMMAs.
__global__ __launch_bounds__(128)
void gemm_kernel(const _Float16* __restrict__ X, const _Float16* __restrict__ Wt,
                 _Float16* __restrict__ Yh, float* __restrict__ Yf, float scale) {
  __shared__ __align__(16) _Float16 Bs[2][64 * 32];   // 2 x 4 KB
  int t = threadIdx.x;
  int l = t & 31, r = l & 15, hb = (l >> 4) & 1;
  int w = t >> 5;                       // wave id 0..3
  int m0 = blockIdx.x * 64 + w * 16;
  int n0 = blockIdx.y * 64;

  // stage one 64x32 weight chunk: 256 x 16B slots, 2 async ops per thread
  auto stage = [&](int kc, int buf) {
#pragma unroll
    for (int s = t; s < 256; s += 128) {
      int col = s >> 2, part = s & 3;
      const _Float16* g = Wt + (size_t)(n0 + col) * C_ + kc * 32 + part * 8;
      unsigned loff = (unsigned)(uintptr_t)&Bs[buf][col * 32 + part * 8];
      async_g2l_b128(loff, g);
    }
  };

  stage(0, 0);
  v8f acc[4] = {};
  const _Float16* arow = X + (size_t)(m0 + r) * C_ + hb * 16;
  for (int kc = 0; kc < KCH_; ++kc) {
    if (kc + 1 < KCH_) {
      stage(kc + 1, (kc + 1) & 1);                       // prefetch next chunk
      asm volatile("s_wait_asynccnt 0x2" ::: "memory");  // chunk kc landed (in-order)
    } else {
      asm volatile("s_wait_asynccnt 0x0" ::: "memory");
    }
    __syncthreads();                                     // all waves' DMA visible
    v16h a = frag_a16(arow + kc * 32);
    const _Float16* bbase = &Bs[kc & 1][hb * 16];
#pragma unroll
    for (int j = 0; j < 4; ++j)
      acc[j] = wmma_f16(a, frag_b16(bbase + (j * 16 + r) * 32), acc[j]);
    __syncthreads();                                     // done reading this buffer
  }
#pragma unroll
  for (int j = 0; j < 4; ++j)
#pragma unroll
    for (int i = 0; i < 8; ++i) {
      size_t row = (size_t)m0 + i + hb * 8;
      size_t col = (size_t)n0 + j * 16 + r;
      float v = acc[j][i] * scale;
      if (Yh) Yh[row * C_ + col] = (_Float16)v;
      else    Yf[row * C_ + col] = v;
    }
}

// ---- 4) V transpose: [B][784][384] -> [B][384][800] (zero pad) ---------
__global__ void vtrans_kernel(const _Float16* __restrict__ Vp, _Float16* __restrict__ Vt) {
  long idx = (long)blockIdx.x * blockDim.x + threadIdx.x;
  long total = (long)B_ * C_ * NKVP_;
  if (idx >= total) return;
  int kk = (int)(idx % NKVP_); long t = idx / NKVP_;
  int c = (int)(t % C_); int b = (int)(t / C_);
  _Float16 v = (_Float16)0.f;
  if (kk < NKV_) v = Vp[((size_t)b * NKV_ + kk) * C_ + c];
  Vt[idx] = v;
}

// ---- 5) attention: per (b, head, 16-query tile), one wave --------------
__global__ __launch_bounds__(32)
void attn_kernel(const _Float16* __restrict__ Qp, const _Float16* __restrict__ Kp,
                 const _Float16* __restrict__ Vt, _Float16* __restrict__ O) {
  __shared__ __align__(16) _Float16 SP[16 * NKVP_];   // scores -> exp probs (25.6 KB)
  int l = threadIdx.x, r = l & 15, hb = (l >> 4) & 1;
  int qt = blockIdx.x;          // 196 query tiles
  int h  = blockIdx.y;          // 6 heads
  int b  = blockIdx.z;          // 8 batches
  size_t qbase = (size_t)b * NQ_ + (size_t)qt * 16;

  // Q A-fragments (K = 64 channels -> 2 chunks), Q already scaled by 1/8
  const _Float16* qrow = Qp + (qbase + r) * C_ + h * HD_ + hb * 16;
  v16h qa0 = frag_a16(qrow);
  v16h qa1 = frag_a16(qrow + 32);

  // scores S = Q * K^T, 49 key tiles of 16
  for (int kt = 0; kt < NKV_ / 16; ++kt) {
    const _Float16* kp = Kp + ((size_t)b * NKV_ + kt * 16 + r) * C_ + h * HD_ + hb * 16;
    v8f s = {};
    s = wmma_f16(qa0, frag_b16(kp), s);
    s = wmma_f16(qa1, frag_b16(kp + 32), s);
#pragma unroll
    for (int i = 0; i < 8; ++i)
      SP[(i + hb * 8) * NKVP_ + kt * 16 + r] = (_Float16)s[i];
  }
  __syncthreads();

  // softmax: row = lane&15, each half-wave scans 392 keys; combine via shfl_xor
  int kbeg = hb * (NKV_ / 2), kend = kbeg + NKV_ / 2;
  float mx = -3.0e38f;
  for (int k = kbeg; k < kend; ++k) mx = fmaxf(mx, (float)SP[r * NKVP_ + k]);
  mx = fmaxf(mx, __shfl_xor(mx, 16, 32));
  float sum = 0.f;
  for (int k = kbeg; k < kend; ++k) {
    float e = __expf((float)SP[r * NKVP_ + k] - mx);
    sum += e;
    SP[r * NKVP_ + k] = (_Float16)e;     // unnormalized prob, in place
  }
  sum += __shfl_xor(sum, 16, 32);
  float inv = 1.0f / sum;
  for (int idx = l; idx < 16 * (NKVP_ - NKV_); idx += 32)
    SP[(idx >> 4) * NKVP_ + NKV_ + (idx & 15)] = (_Float16)0.f;   // zero key pad
  __syncthreads();

  // O_tile = P * V : M=16 queries, N=64 channels, K=800 keys (25 chunks)
  v8f acc[4] = {};
  for (int k0 = 0; k0 < NKVP_; k0 += 32) {
    v16h pa = frag_a16(&SP[r * NKVP_ + k0 + hb * 16]);
#pragma unroll
    for (int j = 0; j < 4; ++j) {
      const _Float16* vp = Vt + ((size_t)b * C_ + h * HD_ + j * 16 + r) * NKVP_
                              + k0 + hb * 16;
      acc[j] = wmma_f16(pa, frag_b16(vp), acc[j]);
    }
  }
  // normalize per query row and store O[b][q][h*64 + c]
#pragma unroll
  for (int i = 0; i < 8; ++i) {
    float rinv = __shfl(inv, i + hb * 8, 32);
    size_t row = qbase + i + hb * 8;
#pragma unroll
    for (int j = 0; j < 4; ++j)
      O[row * C_ + h * HD_ + j * 16 + r] = (_Float16)(acc[j][i] * rinv);
  }
}

// ---- launch ------------------------------------------------------------
extern "C" void kernel_launch(void* const* d_in, const int* in_sizes, int n_in,
                              void* d_out, int out_size, void* d_ws, size_t ws_size,
                              hipStream_t stream) {
  (void)in_sizes; (void)n_in; (void)out_size; (void)ws_size;
  const float* inq  = (const float*)d_in[0];
  const float* inkv = (const float*)d_in[1];
  const float* wout = (const float*)d_in[2];
  const float* qdw = (const float*)d_in[3];
  const float* qsc = (const float*)d_in[4];
  const float* qbi = (const float*)d_in[5];
  const float* qmn = (const float*)d_in[6];
  const float* qvr = (const float*)d_in[7];
  const float* qpw = (const float*)d_in[8];
  const float* kdw = (const float*)d_in[9];
  const float* ksc = (const float*)d_in[10];
  const float* kbi = (const float*)d_in[11];
  const float* kmn = (const float*)d_in[12];
  const float* kvr = (const float*)d_in[13];
  const float* kpw = (const float*)d_in[14];
  const float* vdw = (const float*)d_in[15];
  const float* vsc = (const float*)d_in[16];
  const float* vbi = (const float*)d_in[17];
  const float* vmn = (const float*)d_in[18];
  const float* vvr = (const float*)d_in[19];
  const float* vpw = (const float*)d_in[20];
  float* out = (float*)d_out;

  char* wp = (char*)d_ws;
  auto carve = [&](size_t elts) {
    _Float16* p = (_Float16*)wp;
    wp += (elts * sizeof(_Float16) + 255) & ~(size_t)255;
    return p;
  };
  _Float16* dwq = carve((size_t)MQ_ * C_);
  _Float16* dwk = carve((size_t)MKV_ * C_);
  _Float16* dwv = carve((size_t)MKV_ * C_);
  _Float16* Wq  = carve((size_t)C_ * C_);
  _Float16* Wk  = carve((size_t)C_ * C_);
  _Float16* Wv  = carve((size_t)C_ * C_);
  _Float16* Wo  = carve((size_t)C_ * C_);
  _Float16* Qp  = carve((size_t)MQ_ * C_);
  _Float16* Kp  = carve((size_t)MKV_ * C_);
  _Float16* Vp  = carve((size_t)MKV_ * C_);
  _Float16* Vt  = carve((size_t)B_ * C_ * NKVP_);
  _Float16* O   = dwq;   // dwq is dead after the Q projection; reuse for attn output

  const int thr = 256;
  // depthwise + BN (q: stride 1 pad 1 ; k/v: stride 2 pad 0, SAME => pad_lo=0)
  dwbn_kernel<<<((long)MQ_ * C_ + thr - 1) / thr, thr, 0, stream>>>(
      inq, qdw, qsc, qbi, qmn, qvr, dwq, HQ_, WQ_, HQ_, WQ_, 1, 1);
  dwbn_kernel<<<((long)MKV_ * C_ + thr - 1) / thr, thr, 0, stream>>>(
      inkv, kdw, ksc, kbi, kmn, kvr, dwk, HQ_, WQ_, HK_, WK_, 2, 0);
  dwbn_kernel<<<((long)MKV_ * C_ + thr - 1) / thr, thr, 0, stream>>>(
      inkv, vdw, vsc, vbi, vmn, vvr, dwv, HQ_, WQ_, HK_, WK_, 2, 0);
  // weight transposes to f16 [OC][IC]
  wtrans_kernel<<<(C_ * C_ + thr - 1) / thr, thr, 0, stream>>>(qpw, Wq);
  wtrans_kernel<<<(C_ * C_ + thr - 1) / thr, thr, 0, stream>>>(kpw, Wk);
  wtrans_kernel<<<(C_ * C_ + thr - 1) / thr, thr, 0, stream>>>(vpw, Wv);
  wtrans_kernel<<<(C_ * C_ + thr - 1) / thr, thr, 0, stream>>>(wout, Wo);
  // pointwise projections (Q pre-scaled by 1/sqrt(64)); 64x64 tile per block
  gemm_kernel<<<dim3(MQ_ / 64, C_ / 64), 128, 0, stream>>>(dwq, Wq, Qp, nullptr, 0.125f);
  gemm_kernel<<<dim3(MKV_ / 64, C_ / 64), 128, 0, stream>>>(dwk, Wk, Kp, nullptr, 1.0f);
  gemm_kernel<<<dim3(MKV_ / 64, C_ / 64), 128, 0, stream>>>(dwv, Wv, Vp, nullptr, 1.0f);
  // V -> [B][C][800] channel-major, zero-padded keys
  vtrans_kernel<<<((long)B_ * C_ * NKVP_ + thr - 1) / thr, thr, 0, stream>>>(Vp, Vt);
  // attention
  attn_kernel<<<dim3(NQ_ / 16, NH_, B_), 32, 0, stream>>>(Qp, Kp, Vt, O);
  // output projection -> f32 d_out
  gemm_kernel<<<dim3(MQ_ / 64, C_ / 64), 128, 0, stream>>>(O, Wo, nullptr, out, 1.0f);
}
